// IGMambaModule_Swapped_77300821394065
// MI455X (gfx1250) — matrix operand order repaired and checked by reading, not compiled, not run on hardware
//
#include <hip/hip_runtime.h>
#include <hip/hip_bf16.h>

#define D_MODEL 128
#define D_STATE 16
#define D_INNER 256
#define DT_RANK 8
#define BB 2
#define HH 64
#define WWID 64
#define LL (HH*WWID)
#define ROWS (BB*LL)          // 8192 tokens
#define EPSV 1e-5f

typedef __attribute__((ext_vector_type(16))) _Float16 v16h;
typedef __attribute__((ext_vector_type(8)))  _Float16 v8h;
typedef __attribute__((ext_vector_type(8)))  float    v8f;

// ---------------- WMMA fragment loaders (16x16x32 f16) ----------------
// A (16xK tile, row-major src, leading dim ld):
//   lanes 0-15 : M=lane,    elems 0..7 <- K 0..7,   elems 8..15 <- K 16..23
//   lanes 16-31: M=lane-16, elems 0..7 <- K 8..15,  elems 8..15 <- K 24..31
__device__ __forceinline__ v16h load_fragA(const _Float16* src, int ld, int lane) {
  int m  = lane & 15;
  int hi = lane >> 4;
  const _Float16* p = src + m * ld + hi * 8;
  v8h lo = *(const v8h*)(p);
  v8h up = *(const v8h*)(p + 16);
  v16h a;
#pragma unroll
  for (int i = 0; i < 8; ++i) { a[i] = lo[i]; a[8 + i] = up[i]; }
  return a;
}
// B (Kx16 tile). Weight stored row-major as (Nrows, K): B[k][n] = W[n][k].
//   lanes 0-15 : N=lane,    elems 0..7 <- K 0..7,   elems 8..15 <- K 16..23
//   lanes 16-31: N=lane-16, elems 0..7 <- K 8..15,  elems 8..15 <- K 24..31
__device__ __forceinline__ v16h load_fragB(const _Float16* w, int ldw, int lane) {
  int n  = lane & 15;
  int kb = (lane >> 4) * 8;
  const _Float16* p = w + n * ldw + kb;
  v8h lo = *(const v8h*)(p);
  v8h up = *(const v8h*)(p + 16);
  v16h b;
#pragma unroll
  for (int i = 0; i < 8; ++i) { b[i] = lo[i]; b[8 + i] = up[i]; }
  return b;
}

// ---------------- kernel 0: pack weights to f16 / precompute A ----------------
__global__ void k_pack(const float* in_proj_w, const float* x_proj_w, const float* out_proj_w,
                       const float* A_log, const float* A_b_log,
                       _Float16* wp_in, _Float16* wp_xp, _Float16* wp_out, float* Avals) {
  int i = blockIdx.x * blockDim.x + threadIdx.x;
  if (i < 512 * 128) wp_in[i] = (_Float16)in_proj_w[i];
  if (i < 64 * 256) {
    int r = i / 256, k = i % 256;
    wp_xp[i] = (r < (DT_RANK + 2 * D_STATE)) ? (_Float16)x_proj_w[r * 256 + k] : (_Float16)0.f;
  }
  if (i < 128 * 256) wp_out[i] = (_Float16)out_proj_w[i];
  if (i < 2 * 256 * 16) {
    int dir = i / (256 * 16), j = i % (256 * 16);
    Avals[i] = -expf(dir ? A_b_log[j] : A_log[j]);
  }
}

// ---------------- kernel 1: front (resize, gate, f_low, RMSnorm) ----------------
__global__ void k_front(const float* __restrict__ x, const float* __restrict__ il,
                        const float* __restrict__ gr, const float* __restrict__ iw,
                        const float* __restrict__ ib, const float* __restrict__ rmsw,
                        const float* __restrict__ alpha_p,
                        float* __restrict__ flow, _Float16* __restrict__ hidden) {
  __shared__ float red[128];
  __shared__ float sh_ci, sh_cg, sh_m, sh_inv;
  int blk = blockIdx.x;                 // b*L + pix
  int b = blk / LL, pix = blk % LL;
  int h = pix / WWID, w = pix % WWID;
  int c = threadIdx.x;
  if (c == 0) {
    float sy = h * 0.5f - 0.25f, sx = w * 0.5f - 0.25f;
    int y0 = (int)floorf(sy), x0 = (int)floorf(sx);
    float fy = sy - (float)y0, fx = sx - (float)x0;
    int y0c = min(max(y0, 0), 31), y1c = min(max(y0 + 1, 0), 31);
    int x0c = min(max(x0, 0), 31), x1c = min(max(x0 + 1, 0), 31);
    const float* ibase = il + b * 32 * 32;
    const float* gbase = gr + b * 32 * 32;
    sh_ci = (1.f - fy) * ((1.f - fx) * ibase[y0c * 32 + x0c] + fx * ibase[y0c * 32 + x1c]) +
            fy * ((1.f - fx) * ibase[y1c * 32 + x0c] + fx * ibase[y1c * 32 + x1c]);
    sh_cg = (1.f - fy) * ((1.f - fx) * gbase[y0c * 32 + x0c] + fx * gbase[y0c * 32 + x1c]) +
            fy * ((1.f - fx) * gbase[y1c * 32 + x0c] + fx * gbase[y1c * 32 + x1c]);
  }
  __syncthreads();
  int xidx = ((b * D_MODEL + c) * HH + h) * WWID + w;
  float xv = x[xidx];
  red[c] = xv * iw[c];
  __syncthreads();
  for (int s = 64; s > 0; s >>= 1) { if (c < s) red[c] += red[c + s]; __syncthreads(); }
  if (c == 0) {
    float dot = red[0] + iw[128] * sh_ci + ib[0];
    sh_m = 1.f / (1.f + expf(-dot));
  }
  __syncthreads();
  flow[xidx] = xv * sh_m;
  float fh = xv * (1.f + alpha_p[0] * sh_cg);
  red[c] = fh * fh;
  __syncthreads();
  for (int s = 64; s > 0; s >>= 1) { if (c < s) red[c] += red[c + s]; __syncthreads(); }
  if (c == 0) sh_inv = rsqrtf(red[0] / 128.f + EPSV);
  __syncthreads();
  hidden[blk * D_MODEL + c] = (_Float16)(fh * sh_inv * rmsw[c]);
}

// ---------------- WMMA GEMM: C(MxN,f32) = A(MxK,f16) * W(NxK,f16)^T ----------------
// one wave computes a 16(M) x 64(N) tile; N % 64 == 0, K % 32 == 0, M % 16 == 0
__global__ void k_gemm(const _Float16* __restrict__ A, const _Float16* __restrict__ W,
                       float* __restrict__ C, int M, int N, int K) {
  int wave = (blockIdx.x * blockDim.x + threadIdx.x) >> 5;
  int lane = threadIdx.x & 31;
  int ntiles = N >> 6;
  int mt = wave / ntiles, nt = wave % ntiles;
  int m0 = mt * 16, n0 = nt * 64;
  v8f acc[4] = {};
  for (int k0 = 0; k0 < K; k0 += 32) {
    v16h a = load_fragA(A + (size_t)m0 * K + k0, K, lane);
#pragma unroll
    for (int t = 0; t < 4; ++t) {
      v16h b = load_fragB(W + (size_t)(n0 + t * 16) * K + k0, K, lane);
      acc[t] = __builtin_amdgcn_wmma_f32_16x16x32_f16(false, a, false, b,
                                                      (short)0, acc[t], false, false);
    }
  }
  int mrow = m0 + (lane >> 4) * 8;
  int ncol = n0 + (lane & 15);
#pragma unroll
  for (int t = 0; t < 4; ++t)
#pragma unroll
    for (int r = 0; r < 8; ++r)
      C[(size_t)(mrow + r) * N + ncol + t * 16] = acc[t][r];
}

// GEMM variant with A = yf + yb (both f16)
__global__ void k_gemm_sum(const _Float16* __restrict__ Af, const _Float16* __restrict__ Ab,
                           const _Float16* __restrict__ W, float* __restrict__ C,
                           int M, int N, int K) {
  int wave = (blockIdx.x * blockDim.x + threadIdx.x) >> 5;
  int lane = threadIdx.x & 31;
  int ntiles = N >> 6;
  int mt = wave / ntiles, nt = wave % ntiles;
  int m0 = mt * 16, n0 = nt * 64;
  v8f acc[4] = {};
  for (int k0 = 0; k0 < K; k0 += 32) {
    v16h a0 = load_fragA(Af + (size_t)m0 * K + k0, K, lane);
    v16h a1 = load_fragA(Ab + (size_t)m0 * K + k0, K, lane);
    v16h a = a0 + a1;
#pragma unroll
    for (int t = 0; t < 4; ++t) {
      v16h b = load_fragB(W + (size_t)(n0 + t * 16) * K + k0, K, lane);
      acc[t] = __builtin_amdgcn_wmma_f32_16x16x32_f16(false, a, false, b,
                                                      (short)0, acc[t], false, false);
    }
  }
  int mrow = m0 + (lane >> 4) * 8;
  int ncol = n0 + (lane & 15);
#pragma unroll
  for (int t = 0; t < 4; ++t)
#pragma unroll
    for (int r = 0; r < 8; ++r)
      C[(size_t)(mrow + r) * N + ncol + t * 16] = acc[t][r];
}

// ---------------- kernel: depthwise causal conv (k=4) + SiLU ----------------
// xz rows (ROWS, 512): cols 0..255 = xs, 256..511 = z
__global__ void k_conv(const float* __restrict__ xz, const float* __restrict__ cw,
                       const float* __restrict__ cb,
                       _Float16* __restrict__ xsc16, float* __restrict__ xsc32) {
  int row = blockIdx.x;           // b*L + l
  int d = threadIdx.x;
  int b = row / LL, l = row % LL;
  float s = cb[d];
#pragma unroll
  for (int j = 0; j < 4; ++j) {
    int ll = l - 3 + j;
    if (ll >= 0) s += cw[d * 4 + j] * xz[(size_t)(b * LL + ll) * 512 + d];
  }
  float y = s / (1.f + expf(-s)) * 1.f;         // silu = s*sigmoid(s)
  y = s * (1.f / (1.f + expf(-s)));
  xsc16[(size_t)row * 256 + d] = (_Float16)y;
  xsc32[(size_t)row * 256 + d] = y;
}

// ---------------- kernel: dt_proj (K=8) + softplus ----------------
__global__ void k_dt(const float* __restrict__ xdbl, const float* __restrict__ dtw,
                     const float* __restrict__ dtb, float* __restrict__ delta) {
  int row = blockIdx.x;
  int d = threadIdx.x;
  const float* xd = xdbl + (size_t)row * 64;
  float acc = dtb[d];
#pragma unroll
  for (int r = 0; r < DT_RANK; ++r) acc += xd[r] * dtw[d * DT_RANK + r];
  float sp = (acc > 20.f) ? acc : log1pf(expf(acc));
  delta[(size_t)row * 256 + d] = sp;
}

// ---------------- kernel: bidirectional selective scan ----------------
// one wave per (b,d); lanes 0-15 = fwd states, lanes 16-31 = bwd states
__global__ void k_scan(const float* __restrict__ delta, const float* __restrict__ u32,
                       const float* __restrict__ xz, const float* __restrict__ xdbl,
                       const float* __restrict__ Avals, const float* __restrict__ Dp,
                       _Float16* __restrict__ yf, _Float16* __restrict__ yb) {
  int wave = (blockIdx.x * blockDim.x + threadIdx.x) >> 5;
  int lane = threadIdx.x & 31;
  int b = wave / D_INNER, d = wave % D_INNER;
  int n = lane & 15;
  int dir = lane >> 4;
  float Av = Avals[(dir * D_INNER + d) * D_STATE + n];
  float Dv = Dp[d];
  float h = 0.f;
  for (int i = 0; i < LL; ++i) {
    int l = dir ? (LL - 1 - i) : i;
    size_t row = (size_t)b * LL + l;
    float dl = delta[row * 256 + d];
    float uv = u32[row * 256 + d];
    float zv = xz[row * 512 + 256 + d];
    float Bv = xdbl[row * 64 + DT_RANK + n];
    float Cv = xdbl[row * 64 + DT_RANK + D_STATE + n];
    h = expf(dl * Av) * h + dl * Bv * uv;
    float p = h * Cv;
    p += __shfl_xor(p, 1, 16);
    p += __shfl_xor(p, 2, 16);
    p += __shfl_xor(p, 4, 16);
    p += __shfl_xor(p, 8, 16);
    if (n == 0) {
      float sz = zv * (1.f / (1.f + expf(-zv)));
      float yv = (p + Dv * uv) * sz;
      if (dir == 0) yf[row * 256 + d] = (_Float16)yv;
      else          yb[row * 256 + d] = (_Float16)yv;
    }
  }
}

// ---------------- kernel: layer-norm + residual combine ----------------
__global__ void k_final(const float* __restrict__ mamba, const float* __restrict__ flow,
                        const float* __restrict__ lnw, const float* __restrict__ lnb,
                        const float* __restrict__ gamma_p, float* __restrict__ out) {
  __shared__ float red[128];
  __shared__ float s_mu, s_inv;
  int blk = blockIdx.x;
  int b = blk / LL, pix = blk % LL;
  int h = pix / WWID, w = pix % WWID;
  int c = threadIdx.x;
  float v = mamba[(size_t)blk * 128 + c];
  red[c] = v;
  __syncthreads();
  for (int s = 64; s > 0; s >>= 1) { if (c < s) red[c] += red[c + s]; __syncthreads(); }
  if (c == 0) s_mu = red[0] / 128.f;
  __syncthreads();
  float dv = v - s_mu;
  red[c] = dv * dv;
  __syncthreads();
  for (int s = 64; s > 0; s >>= 1) { if (c < s) red[c] += red[c + s]; __syncthreads(); }
  if (c == 0) s_inv = rsqrtf(red[0] / 128.f + EPSV);
  __syncthreads();
  int oidx = ((b * D_MODEL + c) * HH + h) * WWID + w;
  out[oidx] = dv * s_inv * lnw[c] + lnb[c] + gamma_p[0] * flow[oidx];
}

// ---------------- launch ----------------
static inline size_t align256(size_t x) { return (x + 255) & ~(size_t)255; }

extern "C" void kernel_launch(void* const* d_in, const int* in_sizes, int n_in,
                              void* d_out, int out_size, void* d_ws, size_t ws_size,
                              hipStream_t stream) {
  const float* x_in   = (const float*)d_in[0];
  const float* illum  = (const float*)d_in[1];
  const float* grad   = (const float*)d_in[2];
  const float* illw   = (const float*)d_in[3];
  const float* illb   = (const float*)d_in[4];
  const float* rmsw   = (const float*)d_in[5];
  const float* inprj  = (const float*)d_in[6];
  const float* convw  = (const float*)d_in[7];
  const float* convb  = (const float*)d_in[8];
  const float* xprj   = (const float*)d_in[9];
  const float* dtw    = (const float*)d_in[10];
  const float* dtb    = (const float*)d_in[11];
  const float* A_log  = (const float*)d_in[12];
  const float* Ab_log = (const float*)d_in[13];
  const float* Dparam = (const float*)d_in[14];
  const float* outprj = (const float*)d_in[15];
  const float* lnw    = (const float*)d_in[16];
  const float* lnb    = (const float*)d_in[17];
  const float* alpha  = (const float*)d_in[18];
  const float* gamma  = (const float*)d_in[19];
  float* out = (float*)d_out;

  char* ws = (char*)d_ws;
  size_t off = 0;
  _Float16* hidden = (_Float16*)(ws + off); off = align256(off + (size_t)ROWS * 128 * 2);
  float*    flow   = (float*)(ws + off);    off = align256(off + (size_t)ROWS * 128 * 4);
  float*    xz     = (float*)(ws + off);    off = align256(off + (size_t)ROWS * 512 * 4);
  _Float16* xsc16  = (_Float16*)(ws + off); off = align256(off + (size_t)ROWS * 256 * 2);
  float*    xsc32  = (float*)(ws + off);    off = align256(off + (size_t)ROWS * 256 * 4);
  float*    xdbl   = (float*)(ws + off);    off = align256(off + (size_t)ROWS * 64 * 4);
  float*    delta  = (float*)(ws + off);    off = align256(off + (size_t)ROWS * 256 * 4);
  _Float16* yf     = (_Float16*)(ws + off); off = align256(off + (size_t)ROWS * 256 * 2);
  _Float16* yb     = (_Float16*)(ws + off); off = align256(off + (size_t)ROWS * 256 * 2);
  float*    mamba  = (float*)(ws + off);    off = align256(off + (size_t)ROWS * 128 * 4);
  _Float16* wp_in  = (_Float16*)(ws + off); off = align256(off + (size_t)512 * 128 * 2);
  _Float16* wp_xp  = (_Float16*)(ws + off); off = align256(off + (size_t)64 * 256 * 2);
  _Float16* wp_out = (_Float16*)(ws + off); off = align256(off + (size_t)128 * 256 * 2);
  float*    Avals  = (float*)(ws + off);    off = align256(off + (size_t)2 * 256 * 16 * 4);

  // 0) pack weights
  k_pack<<<256, 256, 0, stream>>>(inprj, xprj, outprj, A_log, Ab_log,
                                  wp_in, wp_xp, wp_out, Avals);
  // 1) front end
  k_front<<<ROWS, 128, 0, stream>>>(x_in, illum, grad, illw, illb, rmsw, alpha,
                                    flow, hidden);
  // 2) in_proj: (8192x128) x (512x128)^T -> xz (8192x512)
  k_gemm<<<(ROWS / 16) * (512 / 64) * 32 / 256, 256, 0, stream>>>(hidden, wp_in, xz,
                                                                  ROWS, 512, 128);
  // 3) causal conv + silu
  k_conv<<<ROWS, 256, 0, stream>>>(xz, convw, convb, xsc16, xsc32);
  // 4) x_proj: (8192x256) x (64x256)^T -> xdbl (8192x64, cols 40..63 zero)
  k_gemm<<<(ROWS / 16) * (64 / 64) * 32 / 256, 256, 0, stream>>>(xsc16, wp_xp, xdbl,
                                                                 ROWS, 64, 256);
  // 5) dt_proj + softplus
  k_dt<<<ROWS, 256, 0, stream>>>(xdbl, dtw, dtb, delta);
  // 6) bidirectional scan: 512 waves
  k_scan<<<(BB * D_INNER * 32) / 256, 256, 0, stream>>>(delta, xsc32, xz, xdbl,
                                                        Avals, Dparam, yf, yb);
  // 7) out_proj: (8192x256) x (128x256)^T -> mamba (8192x128)
  k_gemm_sum<<<(ROWS / 16) * (128 / 64) * 32 / 256, 256, 0, stream>>>(yf, yb, wp_out,
                                                                      mamba, ROWS, 128, 256);
  // 8) layer norm + gamma*f_low
  k_final<<<ROWS, 128, 0, stream>>>(mamba, flow, lnw, lnb, gamma, out);
}
